// MultiOrderWeightedConv1D_44349832299092
// MI455X (gfx1250) — compile-verified, matrix operand order checked by reading, not run on hardware
//
#include <hip/hip_runtime.h>

#define BB 4
#define CC 64
#define OO 64
#define NN 32768
#define TT 4
#define KT 9
#define CKDIM 576      // reordered K-dim: kk = k*64 + c
#define TILE_A 64      // conv n-tile per block
#define TILE_F 64      // fusion n-tile per block
#define INV_SIGMA 20.0f

typedef _Float16 h16;
typedef __attribute__((ext_vector_type(16))) _Float16 v16h;
typedef __attribute__((ext_vector_type(8)))  _Float16 v8h;
typedef __attribute__((ext_vector_type(8)))  float    v8f;

__device__ __forceinline__ v16h mul_frag(v8h lo, v8h hi, h16 w) {
  v16h f;
#pragma unroll
  for (int e = 0; e < 8; ++e) { f[e] = lo[e] * w; f[e + 8] = hi[e] * w; }
  return f;
}

__device__ __forceinline__ v16h cat_frag(v8h lo, v8h hi) {
  v16h f;
#pragma unroll
  for (int e = 0; e < 8; ++e) { f[e] = lo[e]; f[e + 8] = hi[e]; }
  return f;
}

__global__ void k_zero(float* p, int n) {
  int e = blockIdx.x * blockDim.x + threadIdx.x;
  if (e < n) p[e] = 0.f;
}

// Reorder conv_w (T,O,C,K) -> Wr[i][o][kk=k*64+c] f16; fusion_w (O,256) -> f16.
__global__ void k_reorder(const float* __restrict__ conv_w,
                          const float* __restrict__ fusion_w,
                          h16* __restrict__ Wr, h16* __restrict__ Fr) {
  int e = blockIdx.x * 256 + threadIdx.x;
  if (e < TT * OO * CKDIM) {
    int i = e / (OO * CKDIM), r = e % (OO * CKDIM);
    int o = r / CKDIM, kk = r % CKDIM;
    int k = kk >> 6, c = kk & 63;
    Wr[e] = (h16)conv_w[((i * OO + o) * CC + c) * KT + k];
  }
  if (e < OO * TT * OO) Fr[e] = (h16)fusion_w[e];
}

// Per (b, i, 64-col tile): gather + radial weights + 64x576x64 WMMA GEMM,
// store pre-BN conv output and accumulate BN statistics.
// 8 waves = 4 n-tiles x 2 m-halves; each wave: 2 m-tiles, ping-pong prefetch.
__global__ __launch_bounds__(256) void k_conv(
    const float* __restrict__ input, const float* __restrict__ mcoords,
    const int* __restrict__ indices, const h16* __restrict__ Wr,
    float* __restrict__ convpre, float* __restrict__ gsum,
    float* __restrict__ gsq) {
  __shared__ __align__(16) h16 xg[72 * 64];    // gathered input window, f16
  __shared__ float cc[3 * 73];                 // coords window
  __shared__ h16   wgth[64 * KT];              // radial weights, f16
  __shared__ int   idxl[72];
  __shared__ float lsum[64], lsq[64];

  const int b = blockIdx.y, i = blockIdx.z;
  const int n0 = blockIdx.x * TILE_A;
  const int tid = threadIdx.x;

  // phase 1: indices + coords + stat init
  if (tid < 72) {
    int m = n0 - 4 + tid;
    idxl[tid] = (m >= 0 && m < NN) ? indices[((size_t)b * NN + m) * TT + i] : -1;
  }
  if (tid < 3 * 72) {
    int ch = tid / 72, mm = tid % 72;
    int m = n0 - 4 + mm;
    cc[ch * 73 + mm] = (m >= 0 && m < NN)
        ? mcoords[(((size_t)b * 3 + ch) * NN + m) * TT + i] : 0.f;
  }
  if (tid < 64) { lsum[tid] = 0.f; lsq[tid] = 0.f; }
  __syncthreads();

  // phase 2: gathered x window (f16) + weights
  const float* inb = input + (size_t)b * CC * NN;
  for (int e = tid; e < 72 * 64; e += 256) {
    int mm = e >> 6, c = e & 63;
    int jdx = idxl[mm];
    float v = (jdx >= 0) ? inb[(size_t)c * NN + jdx] : 0.f;
    xg[e] = (h16)v;
  }
  for (int e = tid; e < 64 * KT; e += 256) {
    int n = e / KT, k = e % KT;
    float s = 0.f;
#pragma unroll
    for (int ch = 0; ch < 3; ++ch) {
      float d = cc[ch * 73 + n + k] - cc[ch * 73 + n + 4];
      s += d * d;
    }
    wgth[e] = (h16)((s > 0.f) ? fmaxf(1.f - sqrtf(s) * INV_SIGMA, 0.f) : 1.f);
  }
  __syncthreads();

  // phase 3: WMMA GEMM, ping-pong double-buffered fragment loads.
  const int lane = tid & 31;
  const int w = tid >> 5;
  const int j = lane & 15;
  const int hb = (lane < 16) ? 0 : 8;
  const int nt = w >> 1, mh = w & 1;
  const int nl = nt * 16 + j;
  const h16* Wri = Wr + (size_t)i * OO * CKDIM;
  v8f acc[2] = {};

  v8h aloX[2], ahiX[2], aloY[2], ahiY[2];
  v8h bloX, bhiX, bloY, bhiY;
  h16 hwX, hwY;

  auto loadA = [&](int ks, v8h* lo, v8h* hi) {
    int kb = ks * 32 + hb;
#pragma unroll
    for (int t = 0; t < 2; ++t) {
      const h16* ap = Wri + (size_t)((mh * 2 + t) * 16 + j) * CKDIM + kb;
      lo[t] = *(const v8h*)ap;
      hi[t] = *(const v8h*)(ap + 16);
    }
  };
  auto loadB = [&](int ks, v8h& lo, v8h& hi, h16& wv) {
    int k = ks >> 1, c0 = (ks & 1) << 5;
    const h16* bp = &xg[(nl + k) * 64 + c0 + hb];   // m = n + k - 4
    lo = *(const v8h*)bp;
    hi = *(const v8h*)(bp + 16);
    wv = wgth[nl * KT + k];
  };
  auto step = [&](v8h* lo, v8h* hi, v8h bl, v8h bh, h16 wv) {
    v16h bf = mul_frag(bl, bh, wv);
#pragma unroll
    for (int t = 0; t < 2; ++t) {
      v16h af = cat_frag(lo[t], hi[t]);
      acc[t] = __builtin_amdgcn_wmma_f32_16x16x32_f16(
          false, af, false, bf, (short)0, acc[t], false, false);
    }
  };

  loadA(0, aloX, ahiX);
  loadB(0, bloX, bhiX, hwX);
  for (int ks = 0; ks < 16; ks += 2) {
    loadA(ks + 1, aloY, ahiY);
    loadB(ks + 1, bloY, bhiY, hwY);
    step(aloX, ahiX, bloX, bhiX, hwX);
    loadA(ks + 2, aloX, ahiX);
    loadB(ks + 2, bloX, bhiX, hwX);
    step(aloY, ahiY, bloY, bhiY, hwY);
  }
  loadA(17, aloY, ahiY);
  loadB(17, bloY, bhiY, hwY);
  step(aloX, ahiX, bloX, bhiX, hwX);
  step(aloY, ahiY, bloY, bhiY, hwY);

  // phase 4: store pre-BN + fused BN statistics
  float* outp = convpre + (size_t)((i * BB + b) * OO) * NN;
  const int rowoff = (lane < 16) ? 0 : 8;
  const int n = n0 + nl;
#pragma unroll
  for (int t = 0; t < 2; ++t) {
#pragma unroll
    for (int v = 0; v < 8; ++v) {
      int o = (mh * 2 + t) * 16 + v + rowoff;
      float x = acc[t][v];
      outp[(size_t)o * NN + n] = x;
      float s = x, q = x * x;
      for (int msk = 1; msk < 16; msk <<= 1) {
        s += __shfl_xor(s, msk);
        q += __shfl_xor(q, msk);
      }
      if (j == 0) { atomicAdd(&lsum[o], s); atomicAdd(&lsq[o], q); }
    }
  }
  __syncthreads();
  if (tid < 64) {
    atomicAdd(&gsum[i * OO + tid], lsum[tid]);
    atomicAdd(&gsq[i * OO + tid], lsq[tid]);
  }
}

__global__ void k_bnparams(const float* __restrict__ sum,
                           const float* __restrict__ sq,
                           const float* __restrict__ gamma,
                           const float* __restrict__ beta,
                           float* __restrict__ scale, float* __restrict__ shift,
                           int nch) {
  int t = blockIdx.x * blockDim.x + threadIdx.x;
  if (t < nch) {
    const float inv = 1.f / (float)((size_t)BB * NN);
    float m = sum[t] * inv;
    float v = sq[t] * inv - m * m;
    float sc = gamma[t] * rsqrtf(v + 1e-5f);
    scale[t] = sc;
    shift[t] = beta[t] - m * sc;
  }
}

// Fusion: gather BN'd per-order outputs into 256-ch cat tile (f16, LDS),
// 64x256 WMMA GEMM, + bias, store pre-BN y and accumulate y statistics.
__global__ __launch_bounds__(256) void k_fusion(
    const float* __restrict__ convpre, const int* __restrict__ reindices,
    const h16* __restrict__ Fr, const float* __restrict__ scaleA,
    const float* __restrict__ shiftA, const float* __restrict__ fb,
    float* __restrict__ ypre, float* __restrict__ ysum,
    float* __restrict__ ysq) {
  __shared__ __align__(16) h16 cat[TILE_F * 256];
  __shared__ int   ridx[256];
  __shared__ float sA[256], sB[256];
  __shared__ float lsum[64], lsq[64];

  const int b = blockIdx.y;
  const int n0 = blockIdx.x * TILE_F;
  const int tid = threadIdx.x;

  {
    int i = tid >> 6, n = tid & 63;
    ridx[tid] = reindices[((size_t)b * NN + n0 + n) * TT + i];
    sA[tid] = scaleA[tid];
    sB[tid] = shiftA[tid];
  }
  if (tid < 64) { lsum[tid] = 0.f; lsq[tid] = 0.f; }
  __syncthreads();

  for (int e = tid; e < TILE_F * 256; e += 256) {
    int n = e >> 8, c2 = e & 255;
    int i = c2 >> 6, oc = c2 & 63;
    int rn = ridx[i * 64 + n];
    float v = convpre[(size_t)((i * BB + b) * OO + oc) * NN + rn];
    cat[e] = (h16)(v * sA[c2] + sB[c2]);
  }
  __syncthreads();

  const int lane = tid & 31;
  const int w = tid >> 5;
  const int j = lane & 15;
  const int hb = (lane < 16) ? 0 : 8;
  const int nt = w >> 1, mh = w & 1;
  const int nl = nt * 16 + j;
  v8f acc[2] = {};

  v8h aloX[2], ahiX[2], aloY[2], ahiY[2];
  v8h bloX, bhiX, bloY, bhiY;

  auto loadA = [&](int ks, v8h* lo, v8h* hi) {
#pragma unroll
    for (int t = 0; t < 2; ++t) {
      const h16* ap = Fr + (size_t)((mh * 2 + t) * 16 + j) * 256 + ks * 32 + hb;
      lo[t] = *(const v8h*)ap;
      hi[t] = *(const v8h*)(ap + 16);
    }
  };
  auto loadB = [&](int ks, v8h& lo, v8h& hi) {
    const h16* bp = &cat[nl * 256 + ks * 32 + hb];
    lo = *(const v8h*)bp;
    hi = *(const v8h*)(bp + 16);
  };
  auto step = [&](v8h* lo, v8h* hi, v8h bl, v8h bh) {
    v16h bf = cat_frag(bl, bh);
#pragma unroll
    for (int t = 0; t < 2; ++t) {
      v16h af = cat_frag(lo[t], hi[t]);
      acc[t] = __builtin_amdgcn_wmma_f32_16x16x32_f16(
          false, af, false, bf, (short)0, acc[t], false, false);
    }
  };

  loadA(0, aloX, ahiX);
  loadB(0, bloX, bhiX);
  for (int ks = 0; ks < 6; ks += 2) {
    loadA(ks + 1, aloY, ahiY);
    loadB(ks + 1, bloY, bhiY);
    step(aloX, ahiX, bloX, bhiX);
    loadA(ks + 2, aloX, ahiX);
    loadB(ks + 2, bloX, bhiX);
    step(aloY, ahiY, bloY, bhiY);
  }
  loadA(7, aloY, ahiY);
  loadB(7, bloY, bhiY);
  step(aloX, ahiX, bloX, bhiX);
  step(aloY, ahiY, bloY, bhiY);

  const int rowoff = (lane < 16) ? 0 : 8;
  const int n = n0 + nl;
#pragma unroll
  for (int t = 0; t < 2; ++t) {
#pragma unroll
    for (int v = 0; v < 8; ++v) {
      int o = (mh * 2 + t) * 16 + v + rowoff;
      float x = acc[t][v] + fb[o];
      ypre[(size_t)(b * OO + o) * NN + n] = x;
      float s = x, q = x * x;
      for (int msk = 1; msk < 16; msk <<= 1) {
        s += __shfl_xor(s, msk);
        q += __shfl_xor(q, msk);
      }
      if (j == 0) { atomicAdd(&lsum[o], s); atomicAdd(&lsq[o], q); }
    }
  }
  __syncthreads();
  if (tid < 64) {
    atomicAdd(&ysum[tid], lsum[tid]);
    atomicAdd(&ysq[tid], lsq[tid]);
  }
}

__global__ void k_apply(const float* __restrict__ ypre,
                        const float* __restrict__ fscale,
                        const float* __restrict__ fshift,
                        float* __restrict__ out) {
  size_t e = (size_t)blockIdx.x * 256 + threadIdx.x;
  int o = (int)((e >> 15) & 63);   // N = 2^15
  float v = ypre[e] * fscale[o] + fshift[o];
  out[e] = fmaxf(v, 0.f);
}

extern "C" void kernel_launch(void* const* d_in, const int* in_sizes, int n_in,
                              void* d_out, int out_size, void* d_ws,
                              size_t ws_size, hipStream_t stream) {
  (void)in_sizes; (void)n_in; (void)out_size; (void)ws_size;
  const float* input     = (const float*)d_in[0];
  const float* mcoords   = (const float*)d_in[1];
  const int*   indices   = (const int*)d_in[2];
  const int*   reindices = (const int*)d_in[3];
  const float* conv_w    = (const float*)d_in[4];
  const float* bn_gamma  = (const float*)d_in[5];
  const float* bn_beta   = (const float*)d_in[6];
  const float* fusion_w  = (const float*)d_in[7];
  const float* fusion_b  = (const float*)d_in[8];
  const float* fusion_g  = (const float*)d_in[9];
  const float* fusion_bt = (const float*)d_in[10];
  float* out = (float*)d_out;

  float* wsf    = (float*)d_ws;
  float* gsum   = wsf;            // 256
  float* gsq    = wsf + 256;      // 256
  float* scaleA = wsf + 512;      // 256
  float* shiftA = wsf + 768;      // 256
  float* ysum   = wsf + 1024;     // 64
  float* ysq    = wsf + 1088;     // 64
  float* fscale = wsf + 1152;     // 64
  float* fshift = wsf + 1216;     // 64
  float* convpre = wsf + 2048;                          // T*B*O*N f32
  float* ypre    = convpre + (size_t)TT * BB * OO * NN; // B*O*N f32
  h16*   Wr      = (h16*)(ypre + (size_t)BB * OO * NN); // T*O*576 f16
  h16*   Fr      = Wr + (size_t)TT * OO * CKDIM;        // O*256 f16

  k_zero<<<8, 256, 0, stream>>>(wsf, 2048);
  k_reorder<<<(TT * OO * CKDIM + 255) / 256, 256, 0, stream>>>(conv_w, fusion_w,
                                                               Wr, Fr);
  k_conv<<<dim3(NN / TILE_A, BB, TT), 256, 0, stream>>>(input, mcoords, indices,
                                                        Wr, convpre, gsum, gsq);
  k_bnparams<<<1, 256, 0, stream>>>(gsum, gsq, bn_gamma, bn_beta, scaleA,
                                    shiftA, 256);
  k_fusion<<<dim3(NN / TILE_F, BB), 256, 0, stream>>>(convpre, reindices, Fr,
                                                      scaleA, shiftA, fusion_b,
                                                      ypre, ysum, ysq);
  k_bnparams<<<1, 64, 0, stream>>>(ysum, ysq, fusion_g, fusion_bt, fscale,
                                   fshift, 64);
  k_apply<<<(BB * OO * NN) / 256, 256, 0, stream>>>(ypre, fscale, fshift, out);
}
